// FocusedAttentionGroup_46007689674770
// MI455X (gfx1250) — matrix operand order repaired and checked by reading, not compiled, not run on hardware
//
#include <hip/hip_runtime.h>

typedef __attribute__((ext_vector_type(16))) __bf16 v16bf;
typedef __attribute__((ext_vector_type(8)))  __bf16 v8bf;
typedef __attribute__((ext_vector_type(8)))  float  v8f;

#define B_    2
#define NSEQ  2048
#define CDIM  1024
#define HNUM  16
#define HD    64
#define RJ    32                         // R1*R2 == WMMA K for bf16
#define SCALE 0.17677669529663687f       // (R1*R2)^-0.5

static __device__ inline v8f wmma_bf16(v16bf a, v16bf b, v8f c) {
  return __builtin_amdgcn_wmma_f32_16x16x32_bf16(false, a, false, b, (short)0, c,
                                                 false, false);
}

// A-fragment (16x32 bf16, row-major source, ld = row stride handled by caller):
// lane l: row = l&15, holds K = jg..jg+15 with jg = (l>>4)*16.
static __device__ inline v16bf load_a_frag(const __bf16* __restrict__ row, int k, int hi) {
  union { v16bf v; v8bf h[2]; } u;
  const int jg = hi * 16;
  u.h[0] = *(const v8bf*)(row + k + jg);
  u.h[1] = *(const v8bf*)(row + k + 8 + jg);
  return u.v;
}

__global__ void cvt_kernel(const float* __restrict__ s, __bf16* __restrict__ d, int n) {
  int i = blockIdx.x * blockDim.x + threadIdx.x;
  if (i < n) d[i] = (__bf16)s[i];
}

// Fused low-rank projections + Kronecker combine.
// Block: 128 thr (4 waves); each block = 16 tokens; 24 col-tiles (q1|k1|q2|k2 = 384 cols).
__global__ __launch_bounds__(128) void proj_qk_kernel(
    const __bf16* __restrict__ xb,
    const __bf16* __restrict__ wq1, const __bf16* __restrict__ wk1,
    const __bf16* __restrict__ wq2, const __bf16* __restrict__ wk2,
    __bf16* __restrict__ Qc, __bf16* __restrict__ Kc) {
  __shared__ float lds[16 * 384];
  const int lane = threadIdx.x & 31;
  const int wv   = threadIdx.x >> 5;
  const int ln   = lane & 15;
  const int hi   = lane >> 4;
  const int tbase = blockIdx.x * 16;

  const __bf16* xrow = xb + (size_t)(tbase + ln) * CDIM;

  const __bf16* wrow[6];
#pragma unroll
  for (int i = 0; i < 6; ++i) {
    int cc = (wv * 6 + i) * 16 + ln;
    const __bf16* p;
    if (cc < 64)       p = wq1 + (size_t)cc * CDIM;
    else if (cc < 128) p = wk1 + (size_t)(cc - 64) * CDIM;
    else if (cc < 256) p = wq2 + (size_t)(cc - 128) * CDIM;
    else               p = wk2 + (size_t)(cc - 256) * CDIM;
    wrow[i] = p;
  }

  v8f acc[6];
#pragma unroll
  for (int i = 0; i < 6; ++i) acc[i] = (v8f){0, 0, 0, 0, 0, 0, 0, 0};

  for (int k = 0; k < CDIM; k += 32) {
    v16bf a = load_a_frag(xrow, k, hi);
#pragma unroll
    for (int i = 0; i < 6; ++i) {
      v16bf b = *(const v16bf*)(wrow[i] + k + hi * 16);
      acc[i] = wmma_bf16(a, b, acc[i]);
    }
  }

  // D layout: lane l, vgpr r -> row (r + 8*(l>>4)), col (l&15)
#pragma unroll
  for (int i = 0; i < 6; ++i) {
    int col = (wv * 6 + i) * 16 + ln;
#pragma unroll
    for (int r = 0; r < 8; ++r) lds[(r + 8 * hi) * 384 + col] = acc[i][r];
  }
  __syncthreads();

  for (int idx = threadIdx.x; idx < 16 * HNUM * RJ; idx += 128) {
    int row = idx >> 9;
    int rem = idx & 511;
    int h  = rem >> 5;
    int j  = rem & 31;
    int r1 = j >> 3;
    int r2 = j & 7;
    const float* L = lds + row * 384;
    float qv = L[h * 4 + r1] * L[128 + h * 8 + r2] * SCALE;
    float kv = L[64 + h * 4 + r1] * L[256 + h * 8 + r2];
    int t = tbase + row;
    int b = t >> 11;
    int n = t & (NSEQ - 1);
    size_t addr = (((size_t)(b * HNUM + h)) * NSEQ + n) * RJ + j;
    Qc[addr] = (__bf16)qv;
    Kc[addr] = (__bf16)kv;
  }
}

// V projection, output stored transposed: Vt[b][h][d][n] (bf16) so attention's
// V^T A-fragments are contiguous 16B loads.
__global__ __launch_bounds__(128) void projv_kernel(
    const __bf16* __restrict__ xb, const __bf16* __restrict__ wv,
    __bf16* __restrict__ Vt) {
  const int lane = threadIdx.x & 31;
  const int w    = threadIdx.x >> 5;
  const int ln   = lane & 15;
  const int hi   = lane >> 4;
  const int tbase = blockIdx.x * 16;
  const int c = blockIdx.y * 64 + w * 16 + ln;

  const __bf16* xrow = xb + (size_t)(tbase + ln) * CDIM;
  const __bf16* wr   = wv + (size_t)c * CDIM;

  v8f acc = (v8f){0, 0, 0, 0, 0, 0, 0, 0};
  for (int k = 0; k < CDIM; k += 32) {
    v16bf a = load_a_frag(xrow, k, hi);
    v16bf b = *(const v16bf*)(wr + k + hi * 16);
    acc = wmma_bf16(a, b, acc);
  }

  const int h = c >> 6, d = c & 63;
  const int t0 = tbase + 8 * hi;
  const int b = t0 >> 11, n0 = t0 & (NSEQ - 1);
  v8bf pk;
#pragma unroll
  for (int r = 0; r < 8; ++r) pk[r] = (__bf16)acc[r];
  *(v8bf*)(Vt + (((size_t)(b * HNUM + h) * HD + d) * NSEQ + n0)) = pk;
}

// Flash-style causal attention. grid.x = B*H*(N/64); 4 waves, each owns 16 queries.
// Computes S^T = K_tile @ Q^T (per-query reduction = 8 in-lane + 1 xor-16 shuffle),
// then out^T = V^T @ P^T.
__global__ __launch_bounds__(128) void attn_kernel(
    const __bf16* __restrict__ Qc, const __bf16* __restrict__ Kc,
    const __bf16* __restrict__ Vt, __bf16* __restrict__ AO) {
  const int lane = threadIdx.x & 31;
  const int w    = threadIdx.x >> 5;
  const int ln   = lane & 15;
  const int hi   = lane >> 4;
  const int qblk = blockIdx.x & 31;
  const int bh   = blockIdx.x >> 5;
  const int b = bh >> 4, h = bh & 15;
  const int qbase = qblk * 64 + w * 16;

  const __bf16* Qp = Qc + (size_t)bh * NSEQ * RJ;
  const __bf16* Kp = Kc + (size_t)bh * NSEQ * RJ;
  const __bf16* Vp = Vt + (size_t)bh * HD * NSEQ;

  const int q = qbase + ln;
  // Q^T B-fragment (fixed over key loop): lane ln -> query column, K contiguous
  v16bf qf = *(const v16bf*)(Qp + (size_t)q * RJ + hi * 16);

  float m_run = -3.0e38f, l_run = 0.f;
  v8f acc[4];
#pragma unroll
  for (int i = 0; i < 4; ++i) acc[i] = (v8f){0, 0, 0, 0, 0, 0, 0, 0};

  const int qlast = qbase + 15;
  for (int kb = 0; kb <= qlast; kb += 32) {
    v16bf ka0 = load_a_frag(Kp + (size_t)(kb + ln) * RJ, 0, hi);
    v16bf ka1 = load_a_frag(Kp + (size_t)(kb + 16 + ln) * RJ, 0, hi);
    v8f z = (v8f){0, 0, 0, 0, 0, 0, 0, 0};
    v8f S0 = wmma_bf16(ka0, qf, z);   // S^T tile, keys kb..kb+15
    v8f S1 = wmma_bf16(ka1, qf, z);   // keys kb+16..kb+31

    const int k0 = kb + 8 * hi;
    float tmax = -3.0e38f;
#pragma unroll
    for (int r = 0; r < 8; ++r) {
      float s0 = (k0 + r > q)      ? -1.0e30f : S0[r];
      float s1 = (k0 + 16 + r > q) ? -1.0e30f : S1[r];
      S0[r] = s0; S1[r] = s1;
      tmax = fmaxf(tmax, fmaxf(s0, s1));
    }
    tmax = fmaxf(tmax, __shfl_xor(tmax, 16, 32));
    float mnew = fmaxf(m_run, tmax);
    float corr = __expf(m_run - mnew);

    float P0[8], P1[8];
    float psum = 0.f;
#pragma unroll
    for (int r = 0; r < 8; ++r) {
      P0[r] = __expf(S0[r] - mnew);
      P1[r] = __expf(S1[r] - mnew);
      psum += P0[r] + P1[r];
    }
    psum += __shfl_xor(psum, 16, 32);
    l_run = l_run * corr + psum;
    m_run = mnew;
#pragma unroll
    for (int i = 0; i < 4; ++i)
#pragma unroll
      for (int r = 0; r < 8; ++r) acc[i][r] *= corr;

    // D layout of S^T -> B-fragment layout of P^T via one xor-16 shuffle per value
    float o0[8], o1[8];
#pragma unroll
    for (int r = 0; r < 8; ++r) {
      o0[r] = __shfl_xor(P0[r], 16, 32);
      o1[r] = __shfl_xor(P1[r], 16, 32);
    }
    v16bf pf;
#pragma unroll
    for (int r = 0; r < 8; ++r) {
      pf[r]     = (__bf16)(hi ? o1[r] : P0[r]);
      pf[8 + r] = (__bf16)(hi ? P1[r] : o0[r]);
    }

#pragma unroll
    for (int dg = 0; dg < 4; ++dg) {
      const __bf16* vrow = Vp + (size_t)(dg * 16 + ln) * NSEQ;
      v16bf vf = load_a_frag(vrow, kb, hi);   // V^T tile [16 d x 32 keys]
      acc[dg] = wmma_bf16(vf, pf, acc[dg]);
    }
  }

  const float inv = 1.f / l_run;
  const size_t t = (size_t)b * NSEQ + q;
#pragma unroll
  for (int dg = 0; dg < 4; ++dg) {
    v8bf pk;
#pragma unroll
    for (int r = 0; r < 8; ++r) pk[r] = (__bf16)(acc[dg][r] * inv);
    const int d0 = dg * 16 + 8 * hi;
    *(v8bf*)(AO + t * CDIM + h * HD + d0) = pk;
  }
}

// Final projection: out = AO @ Wo^T + bo (f32 output)
__global__ __launch_bounds__(128) void outproj_kernel(
    const __bf16* __restrict__ AO, const __bf16* __restrict__ wo,
    const float* __restrict__ bo, float* __restrict__ out) {
  const int lane = threadIdx.x & 31;
  const int w    = threadIdx.x >> 5;
  const int ln   = lane & 15;
  const int hi   = lane >> 4;
  const int tbase = blockIdx.x * 16;
  const int c = blockIdx.y * 64 + w * 16 + ln;

  const __bf16* arow = AO + (size_t)(tbase + ln) * CDIM;
  const __bf16* wr   = wo + (size_t)c * CDIM;

  v8f acc = (v8f){0, 0, 0, 0, 0, 0, 0, 0};
  for (int k = 0; k < CDIM; k += 32) {
    v16bf a = load_a_frag(arow, k, hi);
    v16bf b = *(const v16bf*)(wr + k + hi * 16);
    acc = wmma_bf16(a, b, acc);
  }

  const float bias = bo[c];
  const int t0 = tbase + 8 * hi;
#pragma unroll
  for (int r = 0; r < 8; ++r) out[(size_t)(t0 + r) * CDIM + c] = acc[r] + bias;
}

extern "C" void kernel_launch(void* const* d_in, const int* in_sizes, int n_in,
                              void* d_out, int out_size, void* d_ws, size_t ws_size,
                              hipStream_t stream) {
  (void)in_sizes; (void)n_in; (void)out_size; (void)ws_size;
  const float* x   = (const float*)d_in[0];
  const float* Wq1 = (const float*)d_in[1];
  const float* Wk1 = (const float*)d_in[2];
  const float* Wq2 = (const float*)d_in[3];
  const float* Wk2 = (const float*)d_in[4];
  const float* Wv  = (const float*)d_in[5];
  const float* Wo  = (const float*)d_in[6];
  const float* bo  = (const float*)d_in[7];
  float* out = (float*)d_out;

  char* ws = (char*)d_ws;
  size_t off = 0;
  auto alloc_bf = [&](size_t elems) -> __bf16* {
    __bf16* p = (__bf16*)(ws + off);
    off += (elems * sizeof(__bf16) + 255) & ~(size_t)255;
    return p;
  };

  const size_t TOK = (size_t)B_ * NSEQ;           // 4096 tokens
  __bf16* xb  = alloc_bf(TOK * CDIM);
  __bf16* wq1 = alloc_bf((size_t)64 * CDIM);
  __bf16* wk1 = alloc_bf((size_t)64 * CDIM);
  __bf16* wq2 = alloc_bf((size_t)128 * CDIM);
  __bf16* wk2 = alloc_bf((size_t)128 * CDIM);
  __bf16* wvb = alloc_bf((size_t)CDIM * CDIM);
  __bf16* wob = alloc_bf((size_t)CDIM * CDIM);
  __bf16* Qc  = alloc_bf((size_t)B_ * HNUM * NSEQ * RJ);
  __bf16* Kc  = alloc_bf((size_t)B_ * HNUM * NSEQ * RJ);
  __bf16* Vt  = alloc_bf((size_t)B_ * HNUM * HD * NSEQ);
  __bf16* AO  = alloc_bf(TOK * CDIM);

  auto cvt = [&](const float* s, __bf16* d, size_t n) {
    cvt_kernel<<<(unsigned)((n + 255) / 256), 256, 0, stream>>>(s, d, (int)n);
  };
  cvt(x,   xb,  TOK * CDIM);
  cvt(Wq1, wq1, (size_t)64 * CDIM);
  cvt(Wk1, wk1, (size_t)64 * CDIM);
  cvt(Wq2, wq2, (size_t)128 * CDIM);
  cvt(Wk2, wk2, (size_t)128 * CDIM);
  cvt(Wv,  wvb, (size_t)CDIM * CDIM);
  cvt(Wo,  wob, (size_t)CDIM * CDIM);

  proj_qk_kernel<<<(unsigned)(TOK / 16), 128, 0, stream>>>(xb, wq1, wk1, wq2, wk2, Qc, Kc);
  projv_kernel<<<dim3((unsigned)(TOK / 16), CDIM / 64), 128, 0, stream>>>(xb, wvb, Vt);
  attn_kernel<<<(unsigned)(B_ * HNUM * (NSEQ / 64)), 128, 0, stream>>>(Qc, Kc, Vt, AO);
  outproj_kernel<<<dim3((unsigned)(TOK / 16), CDIM / 64), 128, 0, stream>>>(AO, wob, bo, out);
}
